// TransformerBlockSTD_35802847379564
// MI455X (gfx1250) — compile-verified, matrix operand order checked by reading, not run on hardware
//
#include <hip/hip_runtime.h>
#include <hip/hip_bf16.h>
#include <stdint.h>

// ---------- vector types ----------
typedef __attribute__((ext_vector_type(4)))  unsigned int u32x4;
typedef __attribute__((ext_vector_type(4)))  float        f32x4;
typedef __attribute__((ext_vector_type(8)))  float        v8f;
typedef __attribute__((ext_vector_type(16))) __bf16       v16bf;

// round-to-nearest-even f32 -> bf16, all math in 32-bit domain
__device__ __forceinline__ unsigned bfround(float f) {
  unsigned u = __builtin_bit_cast(unsigned, f);
  return u + (0x7FFFu + ((u >> 16) & 1u));
}
__device__ __forceinline__ unsigned pack2bf(float a, float b) {
  return (bfround(a) >> 16) | (bfround(b) & 0xFFFF0000u);
}
__device__ __forceinline__ unsigned short f2bf(float f) {
  return (unsigned short)(bfround(f) >> 16);
}

#define BM 128
#define BN 128
#define BK 32
#define LSTR 40   // ushort row stride in LDS (80B: 16B-aligned, conflict-free b128)

// C = scale*(A@B) [+bias] [+residual]; store f32 (OUT_F32) or bf16.
// A: [M,K] row-major (f32 or bf16).  B: [K,N] row-major, or [N,K] bf16 if B_TRANS.
// blockIdx.z batches: strA/strB/strC are per-batch element strides.
template<int A_BF16, int B_BF16, int B_TRANS, int HAS_BIAS, int HAS_RES, int OUT_F32>
__global__ __launch_bounds__(256)
void gemm128(const void* __restrict__ Av, const void* __restrict__ Bv,
             void* __restrict__ Cv,
             const float* __restrict__ bias, const float* __restrict__ residual,
             float scale, int M, int N, int K, int lda, int ldb, int ldc,
             size_t strA, size_t strB, size_t strC)
{
  __shared__ unsigned short As[BM * LSTR];
  __shared__ unsigned short Bs[BN * LSTR];

  const int tid  = threadIdx.x;
  const int lane = tid & 31;
  const int wave = tid >> 5;
  const int wm   = wave >> 1;            // 0..3  (row quadrant, 32 rows)
  const int wn   = wave & 1;             // 0..1  (col half, 64 cols)
  const int m0   = blockIdx.y * BM;
  const int n0   = blockIdx.x * BN;
  const int half = lane >> 4;            // lane-half selects K sub-block
  const int l16  = lane & 15;
  const size_t zA = (size_t)blockIdx.z * strA;
  const size_t zB = (size_t)blockIdx.z * strB;
  const size_t zC = (size_t)blockIdx.z * strC;

  // staging-thread mapping
  const int rA = tid >> 1, hA = tid & 1;   // A / B-trans: row, 16-col half
  const int kB = tid >> 3, sB = tid & 7;   // B no-trans: k row, 16-col segment

  v8f acc[2][4] = {};

  // software-pipeline registers (only the used set survives DCE)
  f32x4 gaf[4]; u32x4 gab[2];
  f32x4 gbf[4]; u32x4 gbb[2];

  auto fetch = [&](int k0) {
    if (A_BF16) {
      const unsigned short* src =
          (const unsigned short*)Av + zA + (size_t)(m0 + rA) * lda + k0 + hA * 16;
      gab[0] = *(const u32x4*)(src);
      gab[1] = *(const u32x4*)(src + 8);
    } else {
      const float* src = (const float*)Av + zA + (size_t)(m0 + rA) * lda + k0 + hA * 16;
      #pragma unroll
      for (int j = 0; j < 4; ++j) gaf[j] = *(const f32x4*)(src + 4 * j);
    }
    if (B_TRANS) {
      const unsigned short* src =
          (const unsigned short*)Bv + zB + (size_t)(n0 + rA) * ldb + k0 + hA * 16;
      gbb[0] = *(const u32x4*)(src);
      gbb[1] = *(const u32x4*)(src + 8);
    } else if (B_BF16) {
      const unsigned short* src =
          (const unsigned short*)Bv + zB + (size_t)(k0 + kB) * ldb + n0 + sB * 16;
      gbb[0] = *(const u32x4*)(src);
      gbb[1] = *(const u32x4*)(src + 8);
    } else {
      const float* src = (const float*)Bv + zB + (size_t)(k0 + kB) * ldb + n0 + sB * 16;
      #pragma unroll
      for (int j = 0; j < 4; ++j) gbf[j] = *(const f32x4*)(src + 4 * j);
    }
  };

  auto commit = [&]() {
    if (A_BF16) {
      *(u32x4*)&As[rA * LSTR + hA * 16]     = gab[0];
      *(u32x4*)&As[rA * LSTR + hA * 16 + 8] = gab[1];
    } else {
      u32x4 p0, p1;                        // pack in 32-bit domain, two b128 stores
      p0[0] = pack2bf(gaf[0].x, gaf[0].y); p0[1] = pack2bf(gaf[0].z, gaf[0].w);
      p0[2] = pack2bf(gaf[1].x, gaf[1].y); p0[3] = pack2bf(gaf[1].z, gaf[1].w);
      p1[0] = pack2bf(gaf[2].x, gaf[2].y); p1[1] = pack2bf(gaf[2].z, gaf[2].w);
      p1[2] = pack2bf(gaf[3].x, gaf[3].y); p1[3] = pack2bf(gaf[3].z, gaf[3].w);
      *(u32x4*)&As[rA * LSTR + hA * 16]     = p0;
      *(u32x4*)&As[rA * LSTR + hA * 16 + 8] = p1;
    }
    if (B_TRANS) {
      *(u32x4*)&Bs[rA * LSTR + hA * 16]     = gbb[0];
      *(u32x4*)&Bs[rA * LSTR + hA * 16 + 8] = gbb[1];
    } else if (B_BF16) {
      union { u32x4 q[2]; unsigned short s[16]; } u;
      u.q[0] = gbb[0]; u.q[1] = gbb[1];
      #pragma unroll
      for (int i = 0; i < 16; ++i) Bs[(sB * 16 + i) * LSTR + kB] = u.s[i];
    } else {
      #pragma unroll
      for (int j = 0; j < 4; ++j) {
        Bs[(sB * 16 + 4 * j + 0) * LSTR + kB] = f2bf(gbf[j].x);
        Bs[(sB * 16 + 4 * j + 1) * LSTR + kB] = f2bf(gbf[j].y);
        Bs[(sB * 16 + 4 * j + 2) * LSTR + kB] = f2bf(gbf[j].z);
        Bs[(sB * 16 + 4 * j + 3) * LSTR + kB] = f2bf(gbf[j].w);
      }
    }
  };

  const int kTiles = K / BK;
  fetch(0);
  for (int kt = 0; kt < kTiles; ++kt) {
    commit();
    __syncthreads();
    if (kt + 1 < kTiles) fetch((kt + 1) * BK);   // hide global latency behind WMMA

    union Frag { u32x4 q[2]; v16bf v; };
    Frag a[2], b[4];
    #pragma unroll
    for (int t = 0; t < 2; ++t) {
      const unsigned short* p = &As[(wm * 32 + t * 16 + l16) * LSTR + half * 8];
      a[t].q[0] = *(const u32x4*)(p);
      a[t].q[1] = *(const u32x4*)(p + 16);
    }
    #pragma unroll
    for (int u = 0; u < 4; ++u) {
      const unsigned short* p = &Bs[(wn * 64 + u * 16 + l16) * LSTR + half * 8];
      b[u].q[0] = *(const u32x4*)(p);
      b[u].q[1] = *(const u32x4*)(p + 16);
    }
    #pragma unroll
    for (int t = 0; t < 2; ++t)
      #pragma unroll
      for (int u = 0; u < 4; ++u)
        acc[t][u] = __builtin_amdgcn_wmma_f32_16x16x32_bf16(
            false, a[t].v, false, b[u].v, (short)0, acc[t][u], false, false);
    __syncthreads();
  }

  // ---- epilogue: C/D layout -> row = r + 8*half, col = lane&15 (branch-free) ----
  float* Cf = (float*)Cv + zC;
  unsigned short* Cb = (unsigned short*)Cv + zC;
  #pragma unroll
  for (int t = 0; t < 2; ++t) {
    const int gmb = m0 + wm * 32 + t * 16 + half * 8;
    #pragma unroll
    for (int u = 0; u < 4; ++u) {
      const int gn = n0 + wn * 64 + u * 16 + l16;
      float bv = 0.0f;
      if (HAS_BIAS) bv = bias[gn];
      #pragma unroll
      for (int r = 0; r < 8; ++r) {
        const size_t off = (size_t)(gmb + r) * ldc + gn;
        float v = acc[t][u][r] * scale + bv;
        if (HAS_RES) v += residual[off];
        if (OUT_F32) Cf[off] = v;
        else         Cb[off] = f2bf(v);
      }
    }
  }
}

// h = relu(xyz @ delta_w1 + delta_b1)   (K=3: plain VALU)
__global__ __launch_bounds__(256)
void posenc_h_kernel(const float* __restrict__ xyz, const float* __restrict__ w1,
                     const float* __restrict__ b1, unsigned short* __restrict__ h)
{
  const size_t idx = (size_t)blockIdx.x * 256 + threadIdx.x;   // 16384*512
  const int j = (int)(idx & 511);
  const size_t i = idx >> 9;
  float v = xyz[i * 3 + 0] * w1[j] + xyz[i * 3 + 1] * w1[512 + j] +
            xyz[i * 3 + 2] * w1[1024 + j] + b1[j];
  h[idx] = f2bf(fmaxf(v, 0.0f));
}

// in-place row softmax over 4096 elements, one 256-thread block per row
__global__ __launch_bounds__(256)
void softmax_kernel(float* __restrict__ attn)
{
  float* row = attn + (size_t)blockIdx.x * 4096;
  const int t = threadIdx.x, lane = t & 31, wv = t >> 5;
  __shared__ float red[8];

  float vals[16];
  float m = -3.402823466e38f;
  #pragma unroll
  for (int i = 0; i < 16; ++i) { vals[i] = row[t + 256 * i]; m = fmaxf(m, vals[i]); }
  #pragma unroll
  for (int o = 16; o; o >>= 1) m = fmaxf(m, __shfl_xor(m, o, 32));
  if (lane == 0) red[wv] = m;
  __syncthreads();
  m = red[0];
  #pragma unroll
  for (int w = 1; w < 8; ++w) m = fmaxf(m, red[w]);

  float s = 0.0f;
  #pragma unroll
  for (int i = 0; i < 16; ++i) { vals[i] = __expf(vals[i] - m); s += vals[i]; }
  #pragma unroll
  for (int o = 16; o; o >>= 1) s += __shfl_xor(s, o, 32);
  __syncthreads();
  if (lane == 0) red[wv] = s;
  __syncthreads();
  s = 0.0f;
  #pragma unroll
  for (int w = 0; w < 8; ++w) s += red[w];
  const float inv = 1.0f / s;
  #pragma unroll
  for (int i = 0; i < 16; ++i) row[t + 256 * i] = vals[i] * inv;
}

extern "C" void kernel_launch(void* const* d_in, const int* in_sizes, int n_in,
                              void* d_out, int out_size, void* d_ws, size_t ws_size,
                              hipStream_t stream)
{
  const float* xyz      = (const float*)d_in[0];
  const float* features = (const float*)d_in[1];
  const float* fc1_w    = (const float*)d_in[2];
  const float* fc1_b    = (const float*)d_in[3];
  const float* w_qs     = (const float*)d_in[4];
  const float* w_ks     = (const float*)d_in[5];
  const float* w_vs     = (const float*)d_in[6];
  const float* dw1      = (const float*)d_in[7];
  const float* db1      = (const float*)d_in[8];
  const float* dw2      = (const float*)d_in[9];
  const float* db2      = (const float*)d_in[10];
  const float* fc2_w    = (const float*)d_in[11];
  const float* fc2_b    = (const float*)d_in[12];

  const int Bb = 4, Nn = 4096, D = 512, P = 128;
  const int BNr = Bb * Nn;                       // 16384 rows total

  float* out_res  = (float*)d_out;                       // [16384,128]
  float* out_attn = (float*)d_out + (size_t)BNr * P;     // [4,4096,4096]

  char* ws = (char*)d_ws;
  const size_t bf_sz = (size_t)BNr * D * sizeof(unsigned short);   // 16 MiB
  unsigned short* x_bf   = (unsigned short*)(ws);
  unsigned short* h_bf   = (unsigned short*)(ws + bf_sz);
  float*          pe_f   = (float*)(ws + 2 * bf_sz);               // f32, 32 MiB
  unsigned short* q_bf   = (unsigned short*)(ws + 2 * bf_sz + (size_t)BNr * D * 4);
  unsigned short* k_bf   = q_bf + (size_t)BNr * D;
  unsigned short* v_bf   = k_bf + (size_t)BNr * D;
  unsigned short* mid_bf = v_bf + (size_t)BNr * D;
  (void)in_sizes; (void)n_in; (void)out_size; (void)ws_size;

  const dim3 blk(256);
  const float sc = 0.04419417382415922f;   // 1/sqrt(512)
  const size_t sQK = (size_t)Nn * D;       // per-batch q/k/v/mid stride
  const size_t sAT = (size_t)Nn * Nn;      // per-batch attn stride

  // x = features @ fc1_w + fc1_b                     [16384,512] bf16
  gemm128<0,0,0, 1,0,0><<<dim3(D / BN, BNr / BM), blk, 0, stream>>>(
      features, fc1_w, x_bf, fc1_b, nullptr, 1.0f, BNr, D, P, P, D, D, 0, 0, 0);

  // h = relu(xyz @ dw1 + db1)                        [16384,512] bf16
  posenc_h_kernel<<<(BNr * D) / 256, blk, 0, stream>>>(xyz, dw1, db1, h_bf);

  // pe = h @ dw2 + db2                               [16384,512] f32
  gemm128<1,0,0, 1,0,1><<<dim3(D / BN, BNr / BM), blk, 0, stream>>>(
      h_bf, dw2, pe_f, db2, nullptr, 1.0f, BNr, D, D, D, D, D, 0, 0, 0);

  // q = x @ w_qs ; k = x @ w_ks                      bf16
  gemm128<1,0,0, 0,0,0><<<dim3(D / BN, BNr / BM), blk, 0, stream>>>(
      x_bf, w_qs, q_bf, nullptr, nullptr, 1.0f, BNr, D, D, D, D, D, 0, 0, 0);
  gemm128<1,0,0, 0,0,0><<<dim3(D / BN, BNr / BM), blk, 0, stream>>>(
      x_bf, w_ks, k_bf, nullptr, nullptr, 1.0f, BNr, D, D, D, D, D, 0, 0, 0);

  // vpe = x @ w_vs + pe                              bf16
  gemm128<1,0,0, 0,1,0><<<dim3(D / BN, BNr / BM), blk, 0, stream>>>(
      x_bf, w_vs, v_bf, nullptr, pe_f, 1.0f, BNr, D, D, D, D, D, 0, 0, 0);

  // S = (q @ k^T) * 1/sqrt(512)  batched over z -> d_out attn region (f32)
  gemm128<1,1,1, 0,0,1><<<dim3(Nn / BN, Nn / BM, Bb), blk, 0, stream>>>(
      q_bf, k_bf, out_attn, nullptr, nullptr, sc,
      Nn, Nn, D, D, D, Nn, sQK, sQK, sAT);

  // softmax rows in place (attn is an output)
  softmax_kernel<<<BNr, blk, 0, stream>>>(out_attn);

  // mid = attn @ vpe  batched over z                 bf16
  gemm128<0,1,0, 0,0,0><<<dim3(D / BN, Nn / BM, Bb), blk, 0, stream>>>(
      out_attn, v_bf, mid_bf, nullptr, nullptr, 1.0f,
      Nn, D, Nn, Nn, D, D, sAT, sQK, sQK);

  // res = mid @ fc2_w + fc2_b + features             f32 -> d_out
  gemm128<1,0,0, 1,1,1><<<dim3(P / BN, BNr / BM), blk, 0, stream>>>(
      mid_bf, fc2_w, out_res, fc2_b, features, 1.0f, BNr, P, D, D, P, P, 0, 0, 0);
}
